// LayerNormLSTM_77481210020038
// MI455X (gfx1250) — matrix operand order repaired and checked by reading, not compiled, not run on hardware
//
#include <hip/hip_runtime.h>

typedef __attribute__((ext_vector_type(16))) __bf16 v16bf;
typedef __attribute__((ext_vector_type(8)))  __bf16 v8bf;
typedef __attribute__((ext_vector_type(8)))  float  v8f;

#define B_ROWS 32768
#define DIM_D  256
#define DIM_H  512
#define DIM_K  768    // D + H
#define DIM_N  2048   // 4H
#define KSTEPS 24     // 768 / 32
#define TM     32     // batch rows per block
#define LDA    776    // padded LDS row stride (bf16 elems): 388 dwords, 388%64=4

// Pack W [K=768, N=2048] f32 row-major into WMMA-fragment-packed bf16:
// wp[((tileN*KSTEPS + ks)*32 + lane)*16 + e] holds B element
//   k = ks*32 + (lane>>4)*16 + e ,  n = tileN*16 + (lane&15)
// so each wave's B fragment load is 1KB contiguous.
__global__ void lnlstm_packW(const float* __restrict__ W, __bf16* __restrict__ wp) {
    int d = blockIdx.x * blockDim.x + threadIdx.x;     // 0 .. 1572863
    int e     = d & 15;
    int lane  = (d >> 4) & 31;
    int ks    = (d >> 9) % KSTEPS;
    int tileN = d / (512 * KSTEPS);
    int k = ks * 32 + (lane >> 4) * 16 + e;
    int n = tileN * 16 + (lane & 15);
    wp[d] = (__bf16)W[k * DIM_N + n];
}

__global__ __launch_bounds__(512)
void lnlstm_main(const float* __restrict__ x,     const float* __restrict__ hprev,
                 const float* __restrict__ cprev, const __bf16* __restrict__ wp,
                 const float* __restrict__ bias,  const float* __restrict__ gamma,
                 const float* __restrict__ beta,  float* __restrict__ out) {
    __shared__ __bf16 sA[TM * LDA];
    __shared__ float  red_s[TM];
    __shared__ float  red_q[TM];

    const int tid  = threadIdx.x;
    const int w    = tid >> 5;        // wave 0..15
    const int lane = tid & 31;
    const int half = lane >> 4;       // 0 | 1
    const int l16  = lane & 15;
    const int r0   = blockIdx.x * TM;

    if (tid < TM) { red_s[tid] = 0.f; red_q[tid] = 0.f; }

    // Stage A = [x | h_prev] rows r0..r0+31 into LDS as bf16 (coalesced reads).
    for (int i = tid; i < TM * DIM_K; i += 512) {
        int row = i / DIM_K;
        int k   = i - row * DIM_K;
        float v = (k < DIM_D) ? x[(r0 + row) * DIM_D + k]
                              : hprev[(r0 + row) * DIM_H + (k - DIM_D)];
        sA[row * LDA + k] = (__bf16)v;
    }
    __syncthreads();

    // Accumulators: acc[rt][t], rt = row tile (16 rows each),
    // t = gate*2 + tt : wave w owns gate-local cols [32w, 32w+32), tile tt of 2.
    v8f acc[2][8];
    #pragma unroll
    for (int rt = 0; rt < 2; ++rt)
        #pragma unroll
        for (int t = 0; t < 8; ++t)
            acc[rt][t] = (v8f){0.f,0.f,0.f,0.f,0.f,0.f,0.f,0.f};

    // Per-wave B base: tileN = g*32 + w*2 + tt -> byte offset from (w*2) base is
    // (g*32+tt)*KSTEPS*1024 bytes, a compile-time constant per t (max 2.38MB,
    // fits the signed 24-bit instruction offset). Per-ks step = 1024 bytes.
    const __bf16* wb = wp + (size_t)((w * 2 * KSTEPS) * 32 + lane) * 16;

    for (int ks = 0; ks < KSTEPS; ++ks, wb += 512) {
        // A fragments: lane l16 -> row, elements 0..15 = k0+16*half+e
        // (contiguous k order, matching packed B)
        v16bf a[2];
        #pragma unroll
        for (int rt = 0; rt < 2; ++rt) {
            const __bf16* ap = &sA[(rt * 16 + l16) * LDA + ks * 32 + 16 * half];
            v8bf lo = *(const v8bf*)(ap);
            v8bf hi = *(const v8bf*)(ap + 8);
            #pragma unroll
            for (int e = 0; e < 8; ++e) { a[rt][e] = lo[e]; a[rt][8 + e] = hi[e]; }
        }
        #pragma unroll
        for (int t = 0; t < 8; ++t) {
            const int g = t >> 1, tt = t & 1;
            // constant immediate offset per t
            v16bf bf = *(const v16bf*)(wb + (g * 32 + tt) * (KSTEPS * 512));
            acc[0][t] = __builtin_amdgcn_wmma_f32_16x16x32_bf16(
                false, a[0], false, bf, (short)0, acc[0][t], false, false);
            acc[1][t] = __builtin_amdgcn_wmma_f32_16x16x32_bf16(
                false, a[1], false, bf, (short)0, acc[1][t], false, false);
        }
    }

    // Add bias -> acc now holds z.
    #pragma unroll
    for (int t = 0; t < 8; ++t) {
        const int g = t >> 1, tt = t & 1;
        float bb = bias[g * DIM_H + w * 32 + tt * 16 + l16];
        #pragma unroll
        for (int rt = 0; rt < 2; ++rt)
            #pragma unroll
            for (int j = 0; j < 8; ++j)
                acc[rt][t][j] += bb;
    }

    // Row statistics: C/D layout -> element j of lane is row rt*16 + j + 8*half,
    // col = <wave cols> + l16. Reduce over this wave's 128 cols, then LDS-atomic
    // across 16 waves.
    #pragma unroll
    for (int rt = 0; rt < 2; ++rt) {
        #pragma unroll
        for (int j = 0; j < 8; ++j) {
            float s = 0.f, q = 0.f;
            #pragma unroll
            for (int t = 0; t < 8; ++t) { float v = acc[rt][t][j]; s += v; q += v * v; }
            #pragma unroll
            for (int m = 1; m <= 8; m <<= 1) {
                s += __shfl_xor(s, m, 32);
                q += __shfl_xor(q, m, 32);
            }
            if (l16 == 0) {
                int r = rt * 16 + j + 8 * half;
                atomicAdd(&red_s[r], s);
                atomicAdd(&red_q[r], q);
            }
        }
    }
    __syncthreads();

    // Epilogue: LayerNorm + LSTM cell. Wave w holds all four gates for
    // gate-local cols nloc = 32w + tt*16 + l16.
    float gam[8], bet[8];
    #pragma unroll
    for (int t = 0; t < 8; ++t) {
        const int g = t >> 1, tt = t & 1;
        int col = g * DIM_H + w * 32 + tt * 16 + l16;
        gam[t] = gamma[col];
        bet[t] = beta[col];
    }
    const float inv = 1.0f / (float)DIM_N;
    float* hout  = out;
    float* coutp = out + (size_t)B_ROWS * DIM_H;
    #pragma unroll
    for (int rt = 0; rt < 2; ++rt) {
        #pragma unroll
        for (int tt = 0; tt < 2; ++tt) {
            const int nloc = w * 32 + tt * 16 + l16;
            #pragma unroll
            for (int j = 0; j < 8; ++j) {
                const int r  = rt * 16 + j + 8 * half;
                float mu  = red_s[r] * inv;
                float var = red_q[r] * inv - mu * mu;
                float rs  = rsqrtf(var + 1e-5f);
                float iz = (acc[rt][0 + tt][j] - mu) * rs * gam[0 + tt] + bet[0 + tt];
                float fz = (acc[rt][2 + tt][j] - mu) * rs * gam[2 + tt] + bet[2 + tt];
                float gz = (acc[rt][4 + tt][j] - mu) * rs * gam[4 + tt] + bet[4 + tt];
                float oz = (acc[rt][6 + tt][j] - mu) * rs * gam[6 + tt] + bet[6 + tt];
                float ig = 1.f / (1.f + __expf(-iz));
                float fg = 1.f / (1.f + __expf(-fz));
                float og = 1.f / (1.f + __expf(-oz));
                float gg = tanhf(gz);
                size_t idx = (size_t)(r0 + r) * DIM_H + nloc;
                float cp = __builtin_nontemporal_load(&cprev[idx]);
                float cn = fg * cp + ig * gg;
                __builtin_nontemporal_store(og * tanhf(cn), &hout[idx]);
                __builtin_nontemporal_store(cn, &coutp[idx]);
            }
        }
    }
}

extern "C" void kernel_launch(void* const* d_in, const int* in_sizes, int n_in,
                              void* d_out, int out_size, void* d_ws, size_t ws_size,
                              hipStream_t stream) {
    const float* x     = (const float*)d_in[0];
    const float* hprev = (const float*)d_in[1];
    const float* cprev = (const float*)d_in[2];
    const float* W     = (const float*)d_in[3];
    const float* b     = (const float*)d_in[4];
    const float* gamma = (const float*)d_in[5];
    const float* beta  = (const float*)d_in[6];
    __bf16* wp = (__bf16*)d_ws;   // 768*2048*2 = 3 MB of scratch

    lnlstm_packW<<<(DIM_K * DIM_N) / 512, 512, 0, stream>>>(W, wp);
    lnlstm_main<<<B_ROWS / TM, 512, 0, stream>>>(x, hprev, cprev, wp, b, gamma, beta,
                                                 (float*)d_out);
}
